// GaussianRenderingLoss_71313636982803
// MI455X (gfx1250) — matrix operand order repaired and checked by reading, not compiled, not run on hardware
//
#include <hip/hip_runtime.h>
#include <hip/hip_bf16.h>

// ---- problem constants (match reference) ----
constexpr int   Bc  = 2;
constexpr int   Np  = 1024;
constexpr int   Hc  = 128;
constexpr int   Wc  = 128;
constexpr int   GDIM = 38;            // 2+1+3+4+1+27
constexpr float C0c  = 0.28209479177387814f;

// Gaussian window (size 7, sigma 1.5): exp(-(d-3)^2/4.5) / 3.6943722
__device__ __forceinline__ float gauss_w(int d) {
    float t = (float)(d - 3);
    float w = __expf(t * t * (-1.0f / 4.5f)) * 0.27068210f;
    return (d >= 0 && d < 7) ? w : 0.0f;
}

typedef __attribute__((ext_vector_type(16))) _Float16 v16h;
typedef __attribute__((ext_vector_type(8)))  float    v8f;

__device__ __forceinline__ void splitf16(float v, _Float16& hi, _Float16& lo) {
    _Float16 h = (_Float16)v;
    hi = h;
    lo = (_Float16)(v - (float)h);
}

// ---------------------------------------------------------------------------
// K0: zero accumulators
// ---------------------------------------------------------------------------
__global__ void init_acc_kernel(float* acc) {
    if (threadIdx.x < 16) acc[threadIdx.x] = 0.0f;
}

// ---------------------------------------------------------------------------
// K1: per-gaussian preprocessing -> unsorted SoA params + opacity entropy
// params p: 0 px, 1 py, 2 a(-.5*inv00), 3 b(-inv01), 4 c(-.5*inv11),
//           5 r, 6 g, 7 b, 8 opac, 9 z
// ---------------------------------------------------------------------------
__global__ void prep_kernel(const float* __restrict__ gauss,
                            const float* __restrict__ intr,
                            float* __restrict__ upar, float* acc) {
    int i = blockIdx.x * blockDim.x + threadIdx.x;
    if (i >= Bc * Np) return;
    int b = i / Np;
    const float* g = gauss + (size_t)i * GDIM;
    const float* K = intr + b * 9;
    float fx = K[0], cx = K[2], fy = K[4], cy = K[5];

    float mx = g[0], my = g[1], mz = g[2];
    float z4 = fmaxf(mz, 1e-4f);
    float px = fx * mx / z4 + cx;
    float py = fy * my / z4 + cy;

    float sx = g[3], sy = g[4], sz = g[5];
    float qw = g[6], qx = g[7], qy = g[8], qz = g[9];
    float R00 = 1.f - 2.f * (qy * qy + qz * qz), R01 = 2.f * (qx * qy - qw * qz), R02 = 2.f * (qx * qz + qw * qy);
    float R10 = 2.f * (qx * qy + qw * qz), R11 = 1.f - 2.f * (qx * qx + qz * qz), R12 = 2.f * (qy * qz - qw * qx);
    float R20 = 2.f * (qx * qz - qw * qy), R21 = 2.f * (qy * qz + qw * qx), R22 = 1.f - 2.f * (qx * qx + qy * qy);
    float A00 = R00 * sx, A01 = R01 * sy, A02 = R02 * sz;
    float A10 = R10 * sx, A11 = R11 * sy, A12 = R12 * sz;
    float A20 = R20 * sx, A21 = R21 * sy, A22 = R22 * sz;
    float Cxx = A00 * A00 + A01 * A01 + A02 * A02;
    float Cxy = A00 * A10 + A01 * A11 + A02 * A12;
    float Cxz = A00 * A20 + A01 * A21 + A02 * A22;
    float Cyy = A10 * A10 + A11 * A11 + A12 * A12;
    float Cyz = A10 * A20 + A11 * A21 + A12 * A22;
    float Czz = A20 * A20 + A21 * A21 + A22 * A22;

    float zc = fmaxf(mz, 1e-6f), zsq = zc * zc;
    float j00 = fx / zc, j02 = -fx * mx / zsq;
    float j11 = fy / zc, j12 = -fy * my / zsq;
    float u0 = j00 * Cxx + j02 * Cxz;
    float u1 = j00 * Cxy + j02 * Cyz;
    float u2 = j00 * Cxz + j02 * Czz;
    float cov00 = u0 * j00 + u2 * j02 + 0.3f;
    float cov01 = u1 * j11 + u2 * j12;
    float v1 = j11 * Cyy + j12 * Cyz;
    float v2 = j11 * Cyz + j12 * Czz;
    float cov11 = v1 * j11 + v2 * j12 + 0.3f;
    float det = fmaxf(cov00 * cov11 - cov01 * cov01, 1e-8f);
    float i00 = cov11 / det, i11 = cov00 / det, i01 = -cov01 / det;

    float cr = fminf(fmaxf(g[11] * C0c + 0.5f, 0.f), 1.f);
    float cg = fminf(fmaxf(g[12] * C0c + 0.5f, 0.f), 1.f);
    float cb = fminf(fmaxf(g[13] * C0c + 0.5f, 0.f), 1.f);
    float op = g[10];

    const int M = Bc * Np;
    upar[0 * M + i] = px;            upar[1 * M + i] = py;
    upar[2 * M + i] = -0.5f * i00;   upar[3 * M + i] = -i01;
    upar[4 * M + i] = -0.5f * i11;
    upar[5 * M + i] = cr;  upar[6 * M + i] = cg;  upar[7 * M + i] = cb;
    upar[8 * M + i] = op;  upar[9 * M + i] = z4;

    float o = fminf(fmaxf(op, 1e-6f), 1.f - 1e-6f);
    float ent = -(o * __logf(o) + (1.f - o) * __logf(1.f - o));
    atomicAdd(&acc[2], ent);
}

// ---------------------------------------------------------------------------
// K2: bitonic sort by depth (per batch, 1024 threads, LDS), gather sorted SoA
// ---------------------------------------------------------------------------
__global__ void sort_kernel(const float* __restrict__ upar, float* __restrict__ spar) {
    __shared__ float skey[Np];
    __shared__ int   sidx[Np];
    const int b = blockIdx.x;
    const int t = threadIdx.x;
    const int M = Bc * Np;
    skey[t] = upar[9 * M + b * Np + t];
    sidx[t] = t;
    __syncthreads();
    for (int k = 2; k <= Np; k <<= 1) {
        for (int j = k >> 1; j > 0; j >>= 1) {
            int ixj = t ^ j;
            if (ixj > t) {
                bool up = ((t & k) == 0);
                float a = skey[t], c = skey[ixj];
                bool sw = up ? (a > c) : (a < c);
                if (sw) {
                    skey[t] = c; skey[ixj] = a;
                    int ti = sidx[t]; sidx[t] = sidx[ixj]; sidx[ixj] = ti;
                }
            }
            __syncthreads();
        }
    }
    int src = sidx[t];
#pragma unroll
    for (int p = 0; p < 10; ++p)
        spar[p * M + b * Np + t] = upar[p * M + b * Np + src];
}

// ---------------------------------------------------------------------------
// K3: tiled alpha-compositing render + fused L1 partial sums.
// Double-buffered chunk staging via CDNA5 async global->LDS loads
// (GLOBAL_LOAD_ASYNC_TO_LDS_B32, ASYNCcnt-tracked, no VGPR round trip).
// ---------------------------------------------------------------------------
__global__ void render_kernel(const float* __restrict__ spar,
                              const float* __restrict__ target_rgb,
                              const float* __restrict__ target_depth,
                              float* __restrict__ rgbImg,
                              float* __restrict__ depthImg,
                              float* acc) {
    const int b  = blockIdx.z;
    const int x  = blockIdx.x * 16 + (threadIdx.x & 15);
    const int y  = blockIdx.y * 16 + (threadIdx.x >> 4);
    const float fxp = (float)x, fyp = (float)y;
    const int M = Bc * Np;
    __shared__ float sh[2][10][256];

    // issue async loads of one chunk into LDS buffer `buf`
    auto issue_chunk = [&](int buf, int chunk) {
#pragma unroll
        for (int p = 0; p < 10; ++p) {
            const float* gp = spar + p * M + b * Np + chunk + threadIdx.x;
            unsigned lds_off = (unsigned)(size_t)(&sh[buf][p][threadIdx.x]);
            asm volatile("global_load_async_to_lds_b32 %0, %1, off"
                         :: "v"(lds_off), "v"(gp) : "memory");
        }
    };

    issue_chunk(0, 0);
    int buf = 0;
    float T = 1.f, rC = 0.f, gC = 0.f, bC = 0.f, dC = 0.f;
    for (int chunk = 0; chunk < Np; chunk += 256) {
        asm volatile("s_wait_asynccnt 0x0" ::: "memory");
        __syncthreads();
        if (chunk + 256 < Np) issue_chunk(buf ^ 1, chunk + 256);
        const float (*cs)[256] = sh[buf];
        for (int n = 0; n < 256; ++n) {
            float dx = fxp - cs[0][n], dy = fyp - cs[1][n];
            float power = dx * (cs[2][n] * dx + cs[3][n] * dy) + cs[4][n] * dy * dy;
            power = fminf(fmaxf(power, -10.f), 0.f);
            float alpha = fminf(__expf(power) * cs[8][n], 0.99f);
            float w = T * alpha;
            rC += w * cs[5][n]; gC += w * cs[6][n]; bC += w * cs[7][n];
            dC += w * cs[9][n];
            T *= (1.f - alpha);
        }
        __syncthreads();
        buf ^= 1;
    }
    rC = fminf(fmaxf(rC, 0.f), 1.f);
    gC = fminf(fmaxf(gC, 0.f), 1.f);
    bC = fminf(fmaxf(bC, 0.f), 1.f);

    const int pix  = y * Wc + x;
    const int base = b * 3 * Hc * Wc;
    rgbImg[base + 0 * Hc * Wc + pix] = rC;
    rgbImg[base + 1 * Hc * Wc + pix] = gC;
    rgbImg[base + 2 * Hc * Wc + pix] = bC;
    depthImg[b * Hc * Wc + pix] = dC;

    float l1 = fabsf(rC - target_rgb[base + 0 * Hc * Wc + pix])
             + fabsf(gC - target_rgb[base + 1 * Hc * Wc + pix])
             + fabsf(bC - target_rgb[base + 2 * Hc * Wc + pix]);
    float l1d = fabsf(dC - target_depth[b * Hc * Wc + pix]);

    __shared__ float redA[256];
    __shared__ float redB[256];
    redA[threadIdx.x] = l1;
    redB[threadIdx.x] = l1d;
    __syncthreads();
    for (int s = 128; s > 0; s >>= 1) {
        if (threadIdx.x < s) {
            redA[threadIdx.x] += redA[threadIdx.x + s];
            redB[threadIdx.x] += redB[threadIdx.x + s];
        }
        __syncthreads();
    }
    if (threadIdx.x == 0) {
        atomicAdd(&acc[0], redA[0]);
        atomicAdd(&acc[1], redB[0]);
    }
}

// ---------------------------------------------------------------------------
// K4: SSIM horizontal pass via WMMA (16x16 tile x banded 32x16 tap matrix).
// One wave per (q, b*c, tile). Data operand split hi/lo f16 (chained WMMAs).
// q: 0=x, 1=y, 2=x*x, 3=y*y, 4=x*y
// ---------------------------------------------------------------------------
__global__ void ssim_h_kernel(const float* __restrict__ rgbImg,
                              const float* __restrict__ target,
                              float* __restrict__ tmp5) {
    const int tile = blockIdx.x & 63;
    const int rest = blockIdx.x >> 6;
    const int bc = rest % 6;
    const int q  = rest / 6;
    const int ty = tile >> 3, tx = tile & 7;
    const int lane = threadIdx.x;
    const int Mrow = lane & 15;
    const int kOffA = (lane < 16) ? 0 : 16;
    const int Ncol  = lane & 15;
    const int kOffB = (lane < 16) ? 0 : 16;

    const float* X = rgbImg + (size_t)bc * Hc * Wc;
    const float* Y = target + (size_t)bc * Hc * Wc;
    const int yrow = ty * 16 + Mrow;

    // A: data, M=y row, K=input-x index (x0-3 .. x0+28, zero-padded)
    v16h Ahi = {}, Alo = {};
#pragma unroll
    for (int e = 0; e < 16; ++e) {
        int K = e + kOffA;
        int xin = tx * 16 - 3 + K;
        float v = 0.f;
        if (K < 22 && xin >= 0 && xin < Wc) {
            int p = yrow * Wc + xin;
            float xv = X[p];
            if (q == 0)      v = xv;
            else if (q == 1) v = Y[p];
            else if (q == 2) v = xv * xv;
            else if (q == 3) { float yv = Y[p]; v = yv * yv; }
            else             v = xv * Y[p];
        }
        _Float16 h, l; splitf16(v, h, l);
        Ahi[e] = h; Alo[e] = l;
    }
    // B: banded taps, K x N, B[k][n] = g[k-n]
    v16h Bw = {};
#pragma unroll
    for (int e = 0; e < 16; ++e) {
        int K = e + kOffB;
        Bw[e] = (_Float16)gauss_w(K - Ncol);
    }
    v8f Cz = {};
    v8f D = __builtin_amdgcn_wmma_f32_16x16x32_f16(false, Alo, false, Bw, (short)0, Cz, false, false);
    D     = __builtin_amdgcn_wmma_f32_16x16x32_f16(false, Ahi, false, Bw, (short)0, D,  false, false);

    float* out = tmp5 + (size_t)q * (6 * Hc * Wc) + (size_t)bc * Hc * Wc;
#pragma unroll
    for (int r = 0; r < 8; ++r) {
        int oy = ty * 16 + r + ((lane < 16) ? 0 : 8);
        int ox = tx * 16 + Ncol;
        out[oy * Wc + ox] = D[r];
    }
}

// ---------------------------------------------------------------------------
// K5: SSIM vertical pass via WMMA (banded 16x32 taps x data 32x16) + reduce.
// One wave per (b*c, tile); 5 quantities, data split hi/lo (2 WMMAs each).
// ---------------------------------------------------------------------------
__global__ void ssim_v_kernel(const float* __restrict__ tmp5, float* acc) {
    const int tile = blockIdx.x & 63;
    const int bc   = blockIdx.x >> 6;
    const int ty = tile >> 3, tx = tile & 7;
    const int lane  = threadIdx.x;
    const int Mrow  = lane & 15;
    const int kOffA = (lane < 16) ? 0 : 16;
    const int Ncol  = lane & 15;
    const int kOffB = (lane < 16) ? 0 : 16;

    // A: banded taps, M=output y, K=input y: A[m][k] = g[k-m]
    v16h Aw = {};
#pragma unroll
    for (int e = 0; e < 16; ++e) {
        int K = e + kOffA;
        Aw[e] = (_Float16)gauss_w(K - Mrow);
    }

    v8f res[5];
#pragma unroll
    for (int qq = 0; qq < 5; ++qq) {
        const float* src = tmp5 + (size_t)qq * (6 * Hc * Wc) + (size_t)bc * Hc * Wc;
        v16h Bhi = {}, Blo = {};
#pragma unroll
        for (int e = 0; e < 16; ++e) {
            int K = e + kOffB;
            int yin = ty * 16 - 3 + K;
            float v = 0.f;
            if (K < 22 && yin >= 0 && yin < Hc)
                v = src[yin * Wc + tx * 16 + Ncol];
            _Float16 h, l; splitf16(v, h, l);
            Bhi[e] = h; Blo[e] = l;
        }
        v8f Cz = {};
        v8f D = __builtin_amdgcn_wmma_f32_16x16x32_f16(false, Aw, false, Blo, (short)0, Cz, false, false);
        D     = __builtin_amdgcn_wmma_f32_16x16x32_f16(false, Aw, false, Bhi, (short)0, D,  false, false);
        res[qq] = D;
    }

    const float C1 = 1e-4f, C2 = 9e-4f;
    float s = 0.f;
#pragma unroll
    for (int r = 0; r < 8; ++r) {
        float mu1 = res[0][r], mu2 = res[1][r];
        float s1  = res[2][r] - mu1 * mu1;
        float s2  = res[3][r] - mu2 * mu2;
        float s12 = res[4][r] - mu1 * mu2;
        s += (2.f * mu1 * mu2 + C1) * (2.f * s12 + C2)
           / ((mu1 * mu1 + mu2 * mu2 + C1) * (s1 + s2 + C2));
    }
    atomicAdd(&acc[3], s);
}

// ---------------------------------------------------------------------------
// K6: combine loss terms
// ---------------------------------------------------------------------------
__global__ void finalize_kernel(const float* acc, float* out) {
    float l1rgb   = acc[0] / (float)(Bc * 3 * Hc * Wc);
    float l1depth = acc[1] / (float)(Bc * Hc * Wc);
    float opacreg = acc[2] / (float)(Bc * Np);
    float ssim    = acc[3] / (float)(Bc * 3 * Hc * Wc);
    out[0] = 0.8f * l1rgb + 0.2f * (1.f - ssim) + 0.5f * l1depth + 0.01f * opacreg;
}

// ---------------------------------------------------------------------------
extern "C" void kernel_launch(void* const* d_in, const int* in_sizes, int n_in,
                              void* d_out, int out_size, void* d_ws, size_t ws_size,
                              hipStream_t stream) {
    (void)in_sizes; (void)n_in; (void)out_size; (void)ws_size;
    const float* gauss = (const float*)d_in[0];
    const float* intr  = (const float*)d_in[1];
    const float* trgb  = (const float*)d_in[2];
    const float* tdep  = (const float*)d_in[3];

    float* ws       = (float*)d_ws;
    float* acc      = ws;                          // 16
    float* upar     = acc + 16;                    // 10 * B * N
    float* spar     = upar + 10 * Bc * Np;         // 10 * B * N
    float* rgbImg   = spar + 10 * Bc * Np;         // B*3*H*W
    float* depthImg = rgbImg + Bc * 3 * Hc * Wc;   // B*H*W
    float* tmp5     = depthImg + Bc * Hc * Wc;     // 5*B*3*H*W

    init_acc_kernel<<<1, 32, 0, stream>>>(acc);
    prep_kernel<<<(Bc * Np + 255) / 256, 256, 0, stream>>>(gauss, intr, upar, acc);
    sort_kernel<<<Bc, Np, 0, stream>>>(upar, spar);
    render_kernel<<<dim3(Wc / 16, Hc / 16, Bc), 256, 0, stream>>>(spar, trgb, tdep, rgbImg, depthImg, acc);
    ssim_h_kernel<<<5 * 6 * 64, 32, 0, stream>>>(rgbImg, trgb, tmp5);
    ssim_v_kernel<<<6 * 64, 32, 0, stream>>>(tmp5, acc);
    finalize_kernel<<<1, 1, 0, stream>>>(acc, (float*)d_out);
}
